// MaskNet_3908420239792
// MI455X (gfx1250) — matrix operand order repaired and checked by reading, not compiled, not run on hardware
//
#include <hip/hip_runtime.h>
#include <stdint.h>

typedef unsigned short u16;
typedef __attribute__((ext_vector_type(16))) __bf16 v16bf;
typedef __attribute__((ext_vector_type(8)))  float  v8f;

constexpr int kB  = 8;
constexpr int kT  = 3900;
constexpr int kC  = 512;   // ENC_DIM
constexpr int kD  = 256;   // DEC_DIM
constexpr int kL  = 300;   // chunks per batch (T / 13)
constexpr int kW  = 26;    // CTX_LEN + CHUNK
constexpr int kCH = 13;    // CHUNK
constexpr float kEPS = 1e-5f;

// Detect CDNA5 async global->LDS copy builtins (compile-safe fallback if absent)
#if defined(__has_builtin)
#if __has_builtin(__builtin_amdgcn_global_load_async_to_lds_b128) && \
    __has_builtin(__builtin_amdgcn_s_wait_asynccnt)
#define USE_ASYNC 1
#endif
#endif
#ifndef USE_ASYNC
#define USE_ASYNC 0
#endif

#if USE_ASYNC
// Builtin signature (from compiler diagnostic): param0 = v4i AS(1)*, param1 = v4i AS(3)*
typedef int v4i_g __attribute__((vector_size(16)));
typedef __attribute__((address_space(1))) v4i_g* as1v4i;
typedef __attribute__((address_space(3))) v4i_g* as3v4i;
#endif

// ---------------- bf16 helpers (storage-only, no __bf16 arithmetic) ---------
__device__ __forceinline__ u16 f2bf(float x) {
  unsigned u = __float_as_uint(x);
  return (u16)((u + 0x7fffu + ((u >> 16) & 1u)) >> 16);   // round-nearest-even
}
__device__ __forceinline__ float bf2f(u16 h) {
  return __uint_as_float(((unsigned)h) << 16);
}

union FragU { v16bf v; unsigned u[8]; };

// A-matrix fragment: 16x32 bf16 (MxK), source row-major [row][k], ld elems.
// lane: m=L&15, half=L>>4; VGPR v holds K = (v&3)*2 + (v>>2)*16 + half*8 (pair)
__device__ __forceinline__ v16bf load_frag_a(const u16* p, int ld, int row0,
                                             int k0, int rows_valid) {
  int lane = threadIdx.x & 31;
  int m = lane & 15, hf = lane >> 4;
  int r = row0 + m;
  bool ok = r < rows_valid;
  const u16* rp = p + (size_t)(ok ? r : 0) * ld + k0 + hf * 8;
  FragU f;
#pragma unroll
  for (int v = 0; v < 8; ++v) {
    int k = ((v & 3) << 1) + ((v >> 2) << 4);
    unsigned val = *(const unsigned*)(rp + k);
    f.u[v] = ok ? val : 0u;
  }
  return f.v;
}

// Unguarded A fragment (all rows known valid, e.g. weight matrices)
__device__ __forceinline__ v16bf load_frag_a_nc(const u16* p, int ld, int row0,
                                                int k0) {
  int lane = threadIdx.x & 31;
  int m = lane & 15, hf = lane >> 4;
  const u16* rp = p + (size_t)(row0 + m) * ld + k0 + hf * 8;
  FragU f;
#pragma unroll
  for (int v = 0; v < 8; ++v) {
    int k = ((v & 3) << 1) + ((v >> 2) << 4);
    f.u[v] = *(const unsigned*)(rp + k);
  }
  return f.v;
}

// B-matrix fragment: 32x16 bf16 (KxN), source stored [n][k] row-major (k contig).
// lane: n=L&15, half=L>>4; VGPR v holds K = (v&3)*2 + (v>>2)*16 + half*8 (pair). No row guard.
__device__ __forceinline__ v16bf load_frag_b_nc(const u16* p, int ld, int row0,
                                                int k0) {
  int lane = threadIdx.x & 31;
  int n = lane & 15, hf = lane >> 4;
  const u16* rp = p + (size_t)(row0 + n) * ld + k0 + hf * 8;
  FragU f;
#pragma unroll
  for (int v = 0; v < 8; ++v)
    f.u[v] = *(const unsigned*)(rp + ((v & 3) << 1) + ((v >> 2) << 4));
  return f.v;
}

__device__ __forceinline__ v8f wmma_bf16(v16bf a, v16bf b, v8f c) {
  return __builtin_amdgcn_wmma_f32_16x16x32_bf16(false, a, false, b,
                                                 (short)0, c, false, false);
}

// ---------------- misc device helpers ---------------------------------------
__device__ __forceinline__ float pe_val(int r, int c) {
  float den = __expf(-9.210340371976184f * (float)(c & ~1) * (1.f / 256.f));
  float a = (float)r * den;
  return (c & 1) ? __cosf(a) : __sinf(a);
}

// ---------------- weight f32 -> bf16 convert --------------------------------
__global__ __launch_bounds__(256) void cvt_kernel(const float* __restrict__ s,
                                                  u16* __restrict__ d, int n) {
  int i = blockIdx.x * 256 + threadIdx.x;
  if (i < n) d[i] = f2bf(s[i]);
}

// ---------------- encoder: depthwise + channel-LN + relu --------------------
// x: [B][512][T] f32 ; out hbf: [B][T][512] bf16  (k-contiguous for WMMA B)
__global__ __launch_bounds__(256) void enc_dw_kernel(
    const float* __restrict__ x, const float* __restrict__ dww,
    const float* __restrict__ dwb, const float* __restrict__ ln1w,
    const float* __restrict__ ln1b, u16* __restrict__ hbf, int d) {
  __shared__ float sH[512 * 16];
  __shared__ float sRed[2 * 512];
  __shared__ float sMu[16], sRs[16];
  int blk = blockIdx.x;
  int tb = blk % 244, b = blk / 244;
  int t0 = tb * 16;
  int tid = threadIdx.x;
  int tl = tid & 15, cs = tid >> 4;
  int t = t0 + tl;
  const float* xb = x + (size_t)b * kC * kT;
  for (int c = cs; c < kC; c += 16) {
    float h = 0.f;
    if (t < kT) {
      float w0 = dww[c * 3], w1 = dww[c * 3 + 1], w2 = dww[c * 3 + 2];
      float v0 = (t >= 2 * d) ? xb[(size_t)c * kT + t - 2 * d] : 0.f;
      float v1 = (t >= d)     ? xb[(size_t)c * kT + t - d]     : 0.f;
      float v2 = xb[(size_t)c * kT + t];
      h = w0 * v0 + w1 * v1 + w2 * v2 + dwb[c];
    }
    sH[c * 16 + tl] = h;
  }
  __syncthreads();
  float s1 = 0.f, s2 = 0.f;
  for (int c = cs * 32; c < cs * 32 + 32; ++c) {
    float v = sH[c * 16 + tl]; s1 += v; s2 += v * v;
  }
  sRed[cs * 16 + tl] = s1; sRed[256 + cs * 16 + tl] = s2;
  __syncthreads();
  if (tid < 16) {
    float a1 = 0.f, a2 = 0.f;
    for (int s = 0; s < 16; ++s) { a1 += sRed[s * 16 + tid]; a2 += sRed[256 + s * 16 + tid]; }
    float mu = a1 * (1.f / 512.f), var = a2 * (1.f / 512.f) - mu * mu;
    sMu[tid] = mu; sRs[tid] = rsqrtf(var + kEPS);
  }
  __syncthreads();
  if (t < kT) {
    for (int c = cs; c < kC; c += 16) {
      float v = (sH[c * 16 + tl] - sMu[tl]) * sRs[tl] * ln1w[c] + ln1b[c];
      hbf[((size_t)b * kT + t) * kC + c] = f2bf(fmaxf(v, 0.f));
    }
  }
}

// ---------------- encoder: pointwise WMMA GEMM + channel-LN + relu + residual
// out[o][t] = sum_c W[o][c]*h[t][c];  M=o(512), N=t(16/block), K=c(512)
__global__ __launch_bounds__(256) void enc_pw_kernel(
    const u16* __restrict__ hbf, const u16* __restrict__ wbf,
    const float* __restrict__ pwb, const float* __restrict__ ln2w,
    const float* __restrict__ ln2b, const float* __restrict__ xin,
    float* __restrict__ xout) {
  __shared__ __align__(16) u16 sB[16 * kC];   // staged activation tile [t][c]
  __shared__ float sOut[512 * 16];
  __shared__ float sRed[2 * 256];
  __shared__ float sMu[16], sRs[16];
  int blk = blockIdx.x;
  int tb = blk % 244, b = blk / 244;
  int t0 = tb * 16;
  int tid = threadIdx.x, lane = tid & 31, wid = tid >> 5;

  // ---- stage 16KB B tile (contiguous rows, clamped at T) into LDS ----
  {
    const u16* hb = hbf + (size_t)b * kT * kC;
#pragma unroll
    for (int it = 0; it < 4; ++it) {
      int q = it * 256 + tid;          // 16-byte chunk id (0..1023)
      int row = q >> 6;                // 64 chunks per 512-elem row
      int col = (q & 63) * 8;
      int gr = t0 + row; if (gr > kT - 1) gr = kT - 1;
      const u16* src = hb + (size_t)gr * kC + col;
      u16* dst = sB + row * kC + col;
#if USE_ASYNC
      __builtin_amdgcn_global_load_async_to_lds_b128(
          (as1v4i)(unsigned long long)(uintptr_t)src,
          (as3v4i)(unsigned)(uintptr_t)dst, 0, 0);
#else
      *(uint4*)dst = *(const uint4*)src;
#endif
    }
#if USE_ASYNC
    __builtin_amdgcn_s_wait_asynccnt(0);
#endif
    __syncthreads();
  }

  v8f acc[4];
#pragma unroll
  for (int mt = 0; mt < 4; ++mt)
#pragma unroll
    for (int v = 0; v < 8; ++v) acc[mt][v] = 0.f;

  // unroll-by-2: two independent fragment sets cover WMMA->VALU WAR hazards
  for (int kk = 0; kk < kC; kk += 64) {
    __builtin_prefetch(wbf + ((size_t)(wid * 64)) * kC + kk + 128, 0, 1);
    v16bf b0 = load_frag_b_nc(sB, kC, 0, kk);
    v16bf b1 = load_frag_b_nc(sB, kC, 0, kk + 32);
    v16bf a0[4], a1[4];
#pragma unroll
    for (int mt = 0; mt < 4; ++mt) {
      a0[mt] = load_frag_a_nc(wbf, kC, wid * 64 + mt * 16, kk);
      a1[mt] = load_frag_a_nc(wbf, kC, wid * 64 + mt * 16, kk + 32);
    }
#pragma unroll
    for (int mt = 0; mt < 4; ++mt) acc[mt] = wmma_bf16(a0[mt], b0, acc[mt]);
#pragma unroll
    for (int mt = 0; mt < 4; ++mt) acc[mt] = wmma_bf16(a1[mt], b1, acc[mt]);
  }

  int n = lane & 15, hf = lane >> 4;
#pragma unroll
  for (int mt = 0; mt < 4; ++mt)
#pragma unroll
    for (int v = 0; v < 8; ++v) {
      int o = wid * 64 + mt * 16 + v + 8 * hf;
      sOut[o * 16 + n] = acc[mt][v] + pwb[o];
    }
  __syncthreads();
  int tl = tid & 15, seg = tid >> 4;
  float s1 = 0.f, s2 = 0.f;
  for (int o = seg * 32; o < seg * 32 + 32; ++o) {
    float v = sOut[o * 16 + tl]; s1 += v; s2 += v * v;
  }
  sRed[seg * 16 + tl] = s1; sRed[256 + seg * 16 + tl] = s2;
  __syncthreads();
  if (tid < 16) {
    float a1 = 0.f, a2 = 0.f;
    for (int s = 0; s < 16; ++s) { a1 += sRed[s * 16 + tid]; a2 += sRed[256 + s * 16 + tid]; }
    float mu = a1 * (1.f / 512.f), var = a2 * (1.f / 512.f) - mu * mu;
    sMu[tid] = mu; sRs[tid] = rsqrtf(var + kEPS);
  }
  __syncthreads();
  int t = t0 + tl;
  if (t < kT) {
    for (int o = seg; o < kC; o += 16) {
      float v = (sOut[o * 16 + tl] - sMu[tl]) * sRs[tl] * ln2w[o] + ln2b[o];
      v = fmaxf(v, 0.f);
      size_t idx = ((size_t)b * kC + o) * kT + t;
      xout[idx] = xin[idx] + v;
    }
  }
}

// ---------------- gin: 512 -> 256 group projections (ep and m0) -------------
__global__ __launch_bounds__(256) void gin_kernel(
    const float* __restrict__ e, const float* __restrict__ l,
    const float* __restrict__ we, const float* __restrict__ be,
    const float* __restrict__ wl, const float* __restrict__ bl,
    float* __restrict__ ep, float* __restrict__ m0) {
  size_t idx = (size_t)blockIdx.x * 256 + threadIdx.x;
  int t = (int)(idx % kT);
  size_t rg = idx / kT;
  int g = (int)(rg % kD);
  int b = (int)(rg / kD);
  size_t e0i = ((size_t)b * kC + 2 * g) * kT + t;
  float e0 = e[e0i], e1 = e[e0i + kT];
  float l0 = l[b * kC + 2 * g], l1 = l[b * kC + 2 * g + 1];
  ep[idx] = fmaxf(we[2 * g] * e0 + we[2 * g + 1] * e1 + be[g], 0.f);
  m0[idx] = fmaxf(wl[2 * g] * l0 * e0 + wl[2 * g + 1] * l1 * e1 + bl[g], 0.f);
}

// ---------------- decoder building blocks -----------------------------------
// D[r][o] = sum_c A[r][c] * W[o][c] + bias[o];  A: LDS bf16 [R][K], W: bf16 [O][K]
// mode 0: f32 -> outF ; mode 1: relu+bf16 -> outB ; mode 2: bf16 -> outB
__device__ __forceinline__ void proj_gemm(const u16* A, int R, int K,
                                          const u16* W, const float* bias,
                                          int O, float* outF, u16* outB,
                                          int ldo, int mode) {
  int tid = threadIdx.x, lane = tid & 31, wid = tid >> 5;
  int Mt = (R + 15) >> 4, Nt = O >> 4;
  int n = lane & 15, hf = lane >> 4;
  for (int tt = wid; tt < Mt * Nt; tt += 8) {
    int mt = tt % Mt, nt = tt / Mt;
    int o = nt * 16 + n;
    float bv = bias[o];
    v8f acc;
#pragma unroll
    for (int v = 0; v < 8; ++v) acc[v] = bv;
    for (int kk = 0; kk < K; kk += 32) {
      v16bf a = load_frag_a(A, K, mt * 16, kk, R);
      v16bf b = load_frag_b_nc(W, K, nt * 16, kk);   // weight rows always valid
      acc = wmma_bf16(a, b, acc);
    }
#pragma unroll
    for (int v = 0; v < 8; ++v) {
      int r = mt * 16 + v + 8 * hf;
      if (r < R) {
        float xv = acc[v];
        if (mode == 0) outF[r * ldo + o] = xv;
        else {
          if (mode == 1) xv = fmaxf(xv, 0.f);
          outB[r * ldo + o] = f2bf(xv);
        }
      }
    }
  }
}

// 8 waves <-> 8 heads; 13 queries x 26 keys, dh=32 (one lane per dim)
__device__ __forceinline__ void attn_heads(const u16* qp, const u16* kp,
                                           const u16* vp, u16* aout) {
  int tid = threadIdx.x, lane = tid & 31, h = tid >> 5;
  const float scale = 0.17677669529663687f;  // 1/sqrt(32)
  for (int qr = 0; qr < kCH; ++qr) {
    float s = -1e30f;
    if (lane < kW) {
      float acc = 0.f;
      const u16* qrow = qp + qr * kD + h * 32;
      const u16* krow = kp + lane * kD + h * 32;
#pragma unroll
      for (int d = 0; d < 32; ++d) acc += bf2f(qrow[d]) * bf2f(krow[d]);
      s = acc * scale;
    }
    float mx = s;
#pragma unroll
    for (int m = 16; m >= 1; m >>= 1) mx = fmaxf(mx, __shfl_xor(mx, m, 32));
    float p = (lane < kW) ? __expf(s - mx) : 0.f;
    float sum = p;
#pragma unroll
    for (int m = 16; m >= 1; m >>= 1) sum += __shfl_xor(sum, m, 32);
    p /= sum;
    float o = 0.f;
    for (int k = 0; k < kW; ++k) {
      float pk = __shfl(p, k, 32);
      o += pk * bf2f(vp[k * kD + h * 32 + lane]);
    }
    aout[qr * kD + h * 32 + lane] = f2bf(o);
  }
}

// LayerNorm(q + add) over 256 channels; q lives in registers (thread = channel)
__device__ __forceinline__ void ln_resid(float* qreg, const float* addv,
                                         const float* w, const float* bb,
                                         float* sred, u16* qbf) {
  int tid = threadIdx.x, lane = tid & 31, wid = tid >> 5;
  for (int r = 0; r < kCH; ++r) {
    float v = qreg[r] + addv[r * kD + tid];
    float s1 = v, s2 = v * v;
#pragma unroll
    for (int m = 16; m >= 1; m >>= 1) {
      s1 += __shfl_xor(s1, m, 32);
      s2 += __shfl_xor(s2, m, 32);
    }
    if (lane == 0) { sred[wid] = s1; sred[8 + wid] = s2; }
    __syncthreads();
    float a1 = 0.f, a2 = 0.f;
#pragma unroll
    for (int i = 0; i < 8; ++i) { a1 += sred[i]; a2 += sred[8 + i]; }
    float mu = a1 * (1.f / 256.f);
    float var = a2 * (1.f / 256.f) - mu * mu;
    float rs = rsqrtf(var + kEPS);
    float o = (v - mu) * rs * w[tid] + bb[tid];
    qreg[r] = o;
    if (qbf) qbf[r * kD + tid] = f2bf(o);
    __syncthreads();
  }
}

// ---------------- decoder tile kernel: one workgroup per (b, chunk) ---------
__global__ __launch_bounds__(256) void dec_kernel(
    const float* __restrict__ m0, const float* __restrict__ ep,
    const u16* __restrict__ sa_in_w, const float* __restrict__ sa_in_b,
    const u16* __restrict__ sa_out_w, const float* __restrict__ sa_out_b,
    const u16* __restrict__ ca_in_w, const float* __restrict__ ca_in_b,
    const u16* __restrict__ ca_out_w, const float* __restrict__ ca_out_b,
    const u16* __restrict__ l1w, const float* __restrict__ l1b,
    const u16* __restrict__ l2w, const float* __restrict__ l2b,
    const float* __restrict__ n1w, const float* __restrict__ n1b,
    const float* __restrict__ n2w, const float* __restrict__ n2b,
    const float* __restrict__ n3w, const float* __restrict__ n3b,
    float* __restrict__ mout) {
  __shared__ char pool[53248];
  __shared__ float sred[16];
  u16*   P0  = (u16*)pool;                 // 13312B: kv_bf (26x256) / ffh (13x512)
  u16*   P1b = (u16*)(pool + 13312);       // kp bf16 (26x256)
  float* P1f = (float*)(pool + 13312);     // proj out f32 (13x256)
  u16*   P2  = (u16*)(pool + 26624);       // vp bf16 (26x256)
  u16*   P3  = (u16*)(pool + 39936);       // qp bf16 (13x256)
  u16*   P4  = (u16*)(pool + 46592);       // attnout_bf / qbf (13x256)

  int tid = threadIdx.x;
  int b = blockIdx.x / kL, j = blockIdx.x % kL;

  float qreg[kCH];
#pragma unroll
  for (int r = 0; r < kCH; ++r) {
    int t = j * kCH + r;
    qreg[r] = m0[((size_t)b * kD + tid) * kT + t] + pe_val(r + kCH, tid);
  }
  for (int idx = tid; idx < kW * kD; idx += 256) {        // tgt_ctx -> bf16
    int r = idx / kD, c = idx % kD;
    int t = j * kCH + r - kCH;
    float v = (t >= 0) ? m0[((size_t)b * kD + c) * kT + t] : 0.f;
    P0[idx] = f2bf(v + pe_val(r, c));
  }
  __syncthreads();

  // ---- self attention ----
  proj_gemm(P0 + kCH * kD, kCH, kD, sa_in_w,              sa_in_b,          kD, nullptr, P3, kD, 2);
  proj_gemm(P0,            kW,  kD, sa_in_w + kD * kD,    sa_in_b + kD,     kD, nullptr, P1b, kD, 2);
  proj_gemm(P0,            kW,  kD, sa_in_w + 2 * kD * kD, sa_in_b + 2 * kD, kD, nullptr, P2, kD, 2);
  __syncthreads();
  attn_heads(P3, P1b, P2, P4);
  __syncthreads();
  proj_gemm(P4, kCH, kD, sa_out_w, sa_out_b, kD, P1f, nullptr, kD, 0);
  __syncthreads();
  ln_resid(qreg, P1f, n1w, n1b, sred, P4);                 // q1 ; qbf -> P4
  __syncthreads();

  // ---- cross attention ----
  for (int idx = tid; idx < kW * kD; idx += 256) {         // mem_ctx -> bf16
    int r = idx / kD, c = idx % kD;
    int t = j * kCH + r - kCH;
    float v = (t >= 0) ? ep[((size_t)b * kD + c) * kT + t] : 0.f;
    P0[idx] = f2bf(v + pe_val(r, c));
  }
  __syncthreads();
  proj_gemm(P4, kCH, kD, ca_in_w,              ca_in_b,          kD, nullptr, P3, kD, 2);
  proj_gemm(P0, kW,  kD, ca_in_w + kD * kD,    ca_in_b + kD,     kD, nullptr, P1b, kD, 2);
  proj_gemm(P0, kW,  kD, ca_in_w + 2 * kD * kD, ca_in_b + 2 * kD, kD, nullptr, P2, kD, 2);
  __syncthreads();
  attn_heads(P3, P1b, P2, P4);                              // qbf(q1) dead
  __syncthreads();
  proj_gemm(P4, kCH, kD, ca_out_w, ca_out_b, kD, P1f, nullptr, kD, 0);
  __syncthreads();
  ln_resid(qreg, P1f, n2w, n2b, sred, P4);                 // q2 ; qbf -> P4
  __syncthreads();

  // ---- feed forward ----
  proj_gemm(P4, kCH, kD, l1w, l1b, 2 * kD, nullptr, P0, 2 * kD, 1); // relu bf16 -> P0
  __syncthreads();
  proj_gemm(P0, kCH, 2 * kD, l2w, l2b, kD, P1f, nullptr, kD, 0);
  __syncthreads();
  ln_resid(qreg, P1f, n3w, n3b, sred, nullptr);
  __syncthreads();

#pragma unroll
  for (int r = 0; r < kCH; ++r) {
    int t = j * kCH + r;
    mout[((size_t)b * kT + t) * kD + tid] = qreg[r];        // [b][t][c]
  }
}

// ---------------- final: out = l*e + relu(pd2e(m)) --------------------------
__global__ __launch_bounds__(256) void final_kernel(
    const float* __restrict__ xw, const float* __restrict__ l,
    const float* __restrict__ pw, const float* __restrict__ pb,
    const float* __restrict__ md, float* __restrict__ out) {
  size_t idx = (size_t)blockIdx.x * 256 + threadIdx.x;
  int t = (int)(idx % kT);
  size_t r = idx / kT;
  int ch = (int)(r % kC);
  int b = (int)(r / kC);
  int g = ch >> 1, jj = ch & 1;
  float mv = md[((size_t)b * kT + t) * kD + g];
  float pv = fmaxf(pw[2 * g + jj] * mv + pb[ch], 0.f);
  out[idx] = l[b * kC + ch] * xw[idx] + pv;
}

// ---------------- host ------------------------------------------------------
extern "C" void kernel_launch(void* const* d_in, const int* in_sizes, int n_in,
                              void* d_out, int out_size, void* d_ws, size_t ws_size,
                              hipStream_t stream) {
  (void)in_sizes; (void)n_in; (void)out_size; (void)ws_size;
  const float* x        = (const float*)d_in[0];
  const float* l        = (const float*)d_in[1];
  const float* dw_w     = (const float*)d_in[4];
  const float* dw_b     = (const float*)d_in[5];
  const float* ln1w     = (const float*)d_in[6];
  const float* ln1b     = (const float*)d_in[7];
  const float* pw_w     = (const float*)d_in[8];
  const float* pw_b     = (const float*)d_in[9];
  const float* ln2w     = (const float*)d_in[10];
  const float* ln2b     = (const float*)d_in[11];
  const float* pe2de_w  = (const float*)d_in[12];
  const float* pe2de_b  = (const float*)d_in[13];
  const float* pe2dl_w  = (const float*)d_in[14];
  const float* pe2dl_b  = (const float*)d_in[15];
  const float* pd2e_w   = (const float*)d_in[16];
  const float* pd2e_b   = (const float*)d_in[17];
  const float* sa_in_w  = (const float*)d_in[18];
  const float* sa_in_b  = (const float*)d_in[19];
  const float* sa_out_w = (const float*)d_in[20];
  const float* sa_out_b = (const float*)d_in[21];
  const float* ca_in_w  = (const float*)d_in[22];
  const float* ca_in_b  = (const float*)d_in[23];
  const float* ca_out_w = (const float*)d_in[24];
  const float* ca_out_b = (const float*)d_in[25];
  const float* l1_w     = (const float*)d_in[26];
  const float* l1_b     = (const float*)d_in[27];
  const float* l2_w     = (const float*)d_in[28];
  const float* l2_b     = (const float*)d_in[29];
  const float* n1w      = (const float*)d_in[30];
  const float* n1b      = (const float*)d_in[31];
  const float* n2w      = (const float*)d_in[32];
  const float* n2b      = (const float*)d_in[33];
  const float* n3w      = (const float*)d_in[34];
  const float* n3b      = (const float*)d_in[35];

  char* ws = (char*)d_ws;
  u16*   pw_bf  = (u16*)(ws + 0);            // 10*512*512 bf16   = 5,242,880
  u16*   sainB  = (u16*)(ws + 5242880);      // 768*256           = 393,216
  u16*   saoutB = (u16*)(ws + 5636096);      // 256*256           = 131,072
  u16*   cainB  = (u16*)(ws + 5767168);      // 768*256           = 393,216
  u16*   caoutB = (u16*)(ws + 6160384);      // 256*256           = 131,072
  u16*   l1B    = (u16*)(ws + 6291456);      // 512*256           = 262,144
  u16*   l2B    = (u16*)(ws + 6553600);      // 256*512           = 262,144
  u16*   hbf    = (u16*)(ws + 6815744);      // [B][T][512] bf16  = 31,948,800
  float* xw     = (float*)(ws + 38764544);   // [B][512][T] f32   = 63,897,600
  float* epb    = (float*)(ws + 102662144);  // [B][256][T] f32   = 31,948,800
  float* m0b    = (float*)(ws + 134610944);  // [B][256][T] f32   = 31,948,800
  float* md     = (float*)(ws + 6815744);    // reuse hbf region: [B][T][256] f32

  cvt_kernel<<<10240, 256, 0, stream>>>(pw_w, pw_bf, 10 * kC * kC);
  cvt_kernel<<<768,   256, 0, stream>>>(sa_in_w, sainB, 3 * kD * kD);
  cvt_kernel<<<256,   256, 0, stream>>>(sa_out_w, saoutB, kD * kD);
  cvt_kernel<<<768,   256, 0, stream>>>(ca_in_w, cainB, 3 * kD * kD);
  cvt_kernel<<<256,   256, 0, stream>>>(ca_out_w, caoutB, kD * kD);
  cvt_kernel<<<512,   256, 0, stream>>>(l1_w, l1B, 2 * kD * kD);
  cvt_kernel<<<512,   256, 0, stream>>>(l2_w, l2B, 2 * kD * kD);

  for (int i = 0; i < 10; ++i) {
    const float* xin = (i == 0) ? x : xw;
    enc_dw_kernel<<<kB * 244, 256, 0, stream>>>(
        xin, dw_w + i * kC * 3, dw_b + i * kC, ln1w + i * kC, ln1b + i * kC,
        hbf, 1 << i);
    enc_pw_kernel<<<kB * 244, 256, 0, stream>>>(
        hbf, pw_bf + (size_t)i * kC * kC, pw_b + i * kC, ln2w + i * kC,
        ln2b + i * kC, xin, xw);
  }

  gin_kernel<<<(kB * kD * kT) / 256, 256, 0, stream>>>(
      xw, l, pe2de_w, pe2de_b, pe2dl_w, pe2dl_b, epb, m0b);

  dec_kernel<<<kB * kL, 256, 0, stream>>>(
      m0b, epb, sainB, sa_in_b, saoutB, sa_out_b, cainB, ca_in_b, caoutB,
      ca_out_b, l1B, l1_b, l2B, l2_b, n1w, n1b, n2w, n2b, n3w, n3b, md);

  final_kernel<<<(kB * kC * kT) / 256, 256, 0, stream>>>(
      xw, l, pd2e_w, pd2e_b, md, (float*)d_out);
}
